// TensorTrainLayer_3805341024421
// MI455X (gfx1250) — compile-verified, matrix operand order checked by reading, not run on hardware
//
#include <hip/hip_runtime.h>

// ---------------------------------------------------------------------------
// TT layer as one bf16 WMMA GEMM:
//   out[B=4096, N=4096] = X[B, K=2048] * W[K, N] + bias[N]
// W materialized from TT cores in fp32 (~0.6 GFLOP), rounded once to bf16,
// stored transposed (Wt[n][k]) so both WMMA operand fragments load as
// contiguous 128-bit reads. GEMM: 256x128 block tile, 8 waves of 64x64,
// double-buffered LDS, one barrier per K-step.
// ---------------------------------------------------------------------------

typedef __attribute__((ext_vector_type(16))) __bf16 v16bf;
typedef __attribute__((ext_vector_type(8)))  __bf16 v8bf;
typedef __attribute__((ext_vector_type(8)))  float  v8f;

#define BATCH 4096
#define KDIM  2048
#define NDIM  4096
#define LDSS  40   // LDS row stride in bf16 elements (80 B, padded vs 64 B)

__device__ __forceinline__ unsigned short f2bf(float f) {
  // round-to-nearest-even f32 -> bf16 (inputs are finite)
  unsigned int u = __float_as_uint(f);
  u += 0x7FFFu + ((u >> 16) & 1u);
  return (unsigned short)(u >> 16);
}

// W1[i0][o0][i1][o1][r2] = sum_r1 core0[i0][o0][r1] * core1[r1][i1][o1][r2]
__global__ __launch_bounds__(256) void tt_build_w1(
    const float* __restrict__ core0, const float* __restrict__ core1,
    float* __restrict__ w1) {
  int t  = blockIdx.x * 256 + threadIdx.x;   // 2^20 elements
  int r2 = t & 31;
  int o1 = (t >> 5) & 15;
  int i1 = (t >> 9) & 15;
  int o0 = (t >> 13) & 15;
  int i0 = t >> 17;
  const float* c0 = core0 + (i0 * 16 + o0) * 32;
  float acc = 0.f;
#pragma unroll
  for (int r1 = 0; r1 < 32; ++r1)
    acc += c0[r1] * core1[((r1 * 16 + i1) * 16 + o1) * 32 + r2];
  w1[t] = acc;
}

// Wt[n][k] (bf16) = sum_r2 W1[i0][o0][i1][o1][r2] * core2[r2][i2][o2]
//   k = i0*256 + i1*16 + i2, n = o0*256 + o1*16 + o2
__global__ __launch_bounds__(256) void tt_build_wt(
    const float* __restrict__ w1, const float* __restrict__ core2,
    unsigned short* __restrict__ wt) {
  int t = blockIdx.x * 256 + threadIdx.x;    // 2048*4096 elements
  int k = t & (KDIM - 1);
  int n = t >> 11;
  int i0 = k >> 8, i1 = (k >> 4) & 15, i2 = k & 15;
  int o0 = n >> 8, o1 = (n >> 4) & 15, o2 = n & 15;
  const float* w1p = w1 + ((((i0 * 16 + o0) * 16 + i1) * 16 + o1) << 5);
  const float* c2p = core2 + i2 * 16 + o2;
  float acc = 0.f;
#pragma unroll
  for (int r2 = 0; r2 < 32; ++r2)
    acc += w1p[r2] * c2p[r2 * 256];
  wt[t] = f2bf(acc);
}

// X fp32 -> bf16 (x flat layout already equals [b][k])
__global__ __launch_bounds__(256) void tt_cvt_x(
    const float* __restrict__ x, unsigned short* __restrict__ xb) {
  int t = blockIdx.x * 256 + threadIdx.x;
  xb[t] = f2bf(x[t]);
}

// Load one 16x32 bf16 operand fragment from an LDS tile (row-major, LDSS
// stride). Lane l<16: row l, K 0-7 & 16-23; lane l>=16: row l-16, K 8-15 &
// 24-31 — per the CDNA5 16-bit A/B VGPR layout.
__device__ __forceinline__ v16bf load_frag(const __bf16* p) {
  v8bf lo = *(const v8bf*)(p);
  v8bf hi = *(const v8bf*)(p + 16);
  return __builtin_shufflevector(lo, hi, 0, 1, 2, 3, 4, 5, 6, 7,
                                 8, 9, 10, 11, 12, 13, 14, 15);
}

// Per-thread global staging: A tile 256x32 (4 chunks), B tile 128x32 (2).
__device__ __forceinline__ void gload(
    const unsigned short* __restrict__ xb, const unsigned short* __restrict__ wt,
    size_t rowM, size_t rowN, int kt, int tid, v8bf (&ra)[4], v8bf (&rb)[2]) {
  const int r  = tid >> 2;
  const int kc = (tid & 3) * 8;
#pragma unroll
  for (int i = 0; i < 4; ++i)
    ra[i] = *(const v8bf*)(xb + (rowM + r + i * 64) * (size_t)KDIM + kt + kc);
#pragma unroll
  for (int i = 0; i < 2; ++i)
    rb[i] = *(const v8bf*)(wt + (rowN + r + i * 64) * (size_t)KDIM + kt + kc);
}

__device__ __forceinline__ void lstore(
    __bf16* __restrict__ la, __bf16* __restrict__ lb, int tid,
    const v8bf (&ra)[4], const v8bf (&rb)[2]) {
  const int r  = tid >> 2;
  const int kc = (tid & 3) * 8;
#pragma unroll
  for (int i = 0; i < 4; ++i)
    *(v8bf*)(la + (r + i * 64) * LDSS + kc) = ra[i];
#pragma unroll
  for (int i = 0; i < 2; ++i)
    *(v8bf*)(lb + (r + i * 64) * LDSS + kc) = rb[i];
}

// C = Xb * Wt^T + bias. Block 256x128; 8 waves (4x2) of 64x64 each.
__global__ __launch_bounds__(256) void tt_gemm(
    const unsigned short* __restrict__ xb, const unsigned short* __restrict__ wt,
    const float* __restrict__ bias, float* __restrict__ out) {
  __shared__ __align__(16) __bf16 ldsA[2][256 * LDSS];  // 2 x 20 KB
  __shared__ __align__(16) __bf16 ldsB[2][128 * LDSS];  // 2 x 10 KB

  const int tid  = threadIdx.x;
  const int lane = tid & 31;
  const int wave = tid >> 5;
  const int wm   = (wave >> 1) * 64;   // 0,64,128,192
  const int wn   = (wave & 1) * 64;    // 0,64
  const int lrow = lane & 15;
  const int lhi  = lane >> 4;
  const size_t rowM = (size_t)blockIdx.x * 256;
  const size_t rowN = (size_t)blockIdx.y * 128;

  v8f zero = {};
  v8f c[4][4];
#pragma unroll
  for (int mf = 0; mf < 4; ++mf)
#pragma unroll
    for (int nf = 0; nf < 4; ++nf) c[mf][nf] = zero;

  v8bf ra[4], rb[2];
  gload(xb, wt, rowM, rowN, 0, tid, ra, rb);
  lstore(ldsA[0], ldsB[0], tid, ra, rb);
  __syncthreads();

  const int KSTEPS = KDIM / 32;  // 64
  for (int ks = 0; ks < KSTEPS; ++ks) {
    const int p = ks & 1;
    const bool more = (ks + 1) < KSTEPS;
    if (more)  // prefetch next tile; WMMA below hides the latency
      gload(xb, wt, rowM, rowN, (ks + 1) * 32, tid, ra, rb);

    const __bf16* La = ldsA[p];
    const __bf16* Lb = ldsB[p];
    v16bf bfr[4];
#pragma unroll
    for (int nf = 0; nf < 4; ++nf)
      bfr[nf] = load_frag(Lb + (wn + nf * 16 + lrow) * LDSS + lhi * 8);
#pragma unroll
    for (int mf = 0; mf < 4; ++mf) {
      v16bf af = load_frag(La + (wm + mf * 16 + lrow) * LDSS + lhi * 8);
#pragma unroll
      for (int nf = 0; nf < 4; ++nf)
        c[mf][nf] = __builtin_amdgcn_wmma_f32_16x16x32_bf16(
            false, af, false, bfr[nf], (short)0, c[mf][nf], false, false);
    }

    if (more) {
      lstore(ldsA[1 - p], ldsB[1 - p], tid, ra, rb);
      __syncthreads();  // one barrier per K-step (double buffered)
    }
  }

  // Epilogue: C/D layout = VGPR j: lanes 0-15 -> M=j, lanes 16-31 -> M=j+8
#pragma unroll
  for (int nf = 0; nf < 4; ++nf) {
    const int col = (int)rowN + wn + nf * 16 + lrow;
    const float bv = bias[col];
#pragma unroll
    for (int mf = 0; mf < 4; ++mf) {
#pragma unroll
      for (int j = 0; j < 8; ++j) {
        const int row = (int)rowM + wm + mf * 16 + lhi * 8 + j;
        out[(size_t)row * NDIM + col] = c[mf][nf][j] + bv;
      }
    }
  }
}

extern "C" void kernel_launch(void* const* d_in, const int* in_sizes, int n_in,
                              void* d_out, int out_size, void* d_ws, size_t ws_size,
                              hipStream_t stream) {
  (void)in_sizes; (void)n_in; (void)out_size; (void)ws_size;
  const float* x     = (const float*)d_in[0];
  const float* core0 = (const float*)d_in[1];
  const float* core1 = (const float*)d_in[2];
  const float* core2 = (const float*)d_in[3];
  const float* bias  = (const float*)d_in[4];
  float* out = (float*)d_out;

  char* ws = (char*)d_ws;
  float*          w1 = (float*)ws;                           //  4 MB fp32 temp
  unsigned short* wt = (unsigned short*)(ws + (4u << 20));   // 16 MB bf16 Wt[n][k]
  unsigned short* xb = (unsigned short*)(ws + (20u << 20));  // 16 MB bf16 X[b][k]

  tt_build_w1<<<(1 << 20) / 256, 256, 0, stream>>>(core0, core1, w1);
  tt_build_wt<<<(KDIM * NDIM) / 256, 256, 0, stream>>>(w1, core2, wt);
  tt_cvt_x  <<<(BATCH * KDIM) / 256, 256, 0, stream>>>(x, xb);

  dim3 grid(BATCH / 256, NDIM / 128);
  tt_gemm<<<grid, dim3(256), 0, stream>>>(xb, wt, bias, out);
}
